// RNN_16587163698009
// MI455X (gfx1250) — matrix operand order repaired and checked by reading, not compiled, not run on hardware
//
#include <hip/hip_runtime.h>
#include <cmath>

#define RNN_N 8
#define RNN_L 4096
#define RNN_C 256
#define RNN_H 256
#define HPAD  (RNN_H + 8)   // pad LDS rows: 528B stride -> rotates banks, kills 16-way conflicts

typedef __bf16 bf16_t;
typedef __attribute__((ext_vector_type(16))) __bf16 v16bf;
typedef __attribute__((ext_vector_type(8)))  __bf16 v8bf;
typedef __attribute__((ext_vector_type(8)))  float  v8f;

// ---- WMMA fragment loaders (bf16, 16x16x32) --------------------------------
// A-matrix 16x32 (MxK), lane m = lane&15, half = lane>>4:
//   element e in [0,8)  -> K = kc*32 + half*8 + e
//   element e in [8,16) -> K = kc*32 + 16 + half*8 + (e-8)
__device__ __forceinline__ v16bf fragA_from_f32(const float* __restrict__ rowbase,
                                                int kc, int half) {
  const int k0 = kc * 32 + half * 8;
  const float4 f0 = *reinterpret_cast<const float4*>(rowbase + k0);
  const float4 f1 = *reinterpret_cast<const float4*>(rowbase + k0 + 4);
  const float4 f2 = *reinterpret_cast<const float4*>(rowbase + k0 + 16);
  const float4 f3 = *reinterpret_cast<const float4*>(rowbase + k0 + 20);
  v16bf a;
  a[0]=(bf16_t)f0.x;  a[1]=(bf16_t)f0.y;  a[2]=(bf16_t)f0.z;  a[3]=(bf16_t)f0.w;
  a[4]=(bf16_t)f1.x;  a[5]=(bf16_t)f1.y;  a[6]=(bf16_t)f1.z;  a[7]=(bf16_t)f1.w;
  a[8]=(bf16_t)f2.x;  a[9]=(bf16_t)f2.y;  a[10]=(bf16_t)f2.z; a[11]=(bf16_t)f2.w;
  a[12]=(bf16_t)f3.x; a[13]=(bf16_t)f3.y; a[14]=(bf16_t)f3.z; a[15]=(bf16_t)f3.w;
  return a;
}

// B-matrix 32x16 (KxN), lane holds column n = lane&15; element e in [0,8) -> K = kc*32 + half*8 + e,
// element e in [8,16) -> K = kc*32 + 16 + half*8 + (e-8) (same K order as the A fragment).
__device__ __forceinline__ v16bf fragB_from_f32(const float* __restrict__ colbase,
                                                int kc, int half) {
  const int k0 = kc * 32 + half * 8;
  const float4 f0 = *reinterpret_cast<const float4*>(colbase + k0);
  const float4 f1 = *reinterpret_cast<const float4*>(colbase + k0 + 4);
  const float4 f2 = *reinterpret_cast<const float4*>(colbase + k0 + 16);
  const float4 f3 = *reinterpret_cast<const float4*>(colbase + k0 + 20);
  v16bf b;
  b[0]=(bf16_t)f0.x;  b[1]=(bf16_t)f0.y;  b[2]=(bf16_t)f0.z;  b[3]=(bf16_t)f0.w;
  b[4]=(bf16_t)f1.x;  b[5]=(bf16_t)f1.y;  b[6]=(bf16_t)f1.z;  b[7]=(bf16_t)f1.w;
  b[8]=(bf16_t)f2.x;  b[9]=(bf16_t)f2.y;  b[10]=(bf16_t)f2.z; b[11]=(bf16_t)f2.w;
  b[12]=(bf16_t)f3.x; b[13]=(bf16_t)f3.y; b[14]=(bf16_t)f3.z; b[15]=(bf16_t)f3.w;
  return b;
}

// A-fragment straight out of a bf16 LDS row (two ds_load_b128 per fragment).
__device__ __forceinline__ v16bf fragA_from_lds(const bf16_t* rowbase, int kc, int half) {
  const int k0 = kc * 32 + half * 8;
  v8bf lo = *reinterpret_cast<const v8bf*>(rowbase + k0);
  v8bf hi = *reinterpret_cast<const v8bf*>(rowbase + k0 + 16);
  return __builtin_shufflevector(lo, hi, 0,1,2,3,4,5,6,7,8,9,10,11,12,13,14,15);
}

// ---- Phase 1: pre[n,l,h] = x @ W_ih^T + b_ih  (one big WMMA GEMM) ----------
// 8 waves / block, each wave owns one 16x16 output tile; K=256 = 8 bf16 WMMAs.
__global__ __launch_bounds__(256)
void rnn_inproj_wmma(const float* __restrict__ x,
                     const float* __restrict__ W_ih,
                     const float* __restrict__ b_ih,
                     float* __restrict__ pre) {
  const int lane = threadIdx.x & 31;
  const int wave = threadIdx.x >> 5;        // 0..7
  const int half = lane >> 4;
  const int l16  = lane & 15;
  const int mt   = blockIdx.x * 8 + wave;   // 0..2047 (M = N*L = 32768 rows)
  const int nt   = blockIdx.y;              // 0..15   (H/16 tiles)
  const int col  = nt * 16 + l16;

  // Weight tile: B[k][n] = W_ih[n][k] -> read row `col` of W_ih
  v16bf B[8];
  #pragma unroll
  for (int kc = 0; kc < 8; ++kc)
    B[kc] = fragB_from_f32(W_ih + (size_t)col * RNN_C, kc, half);

  const float* xrow = x + (size_t)(mt * 16 + l16) * RNN_C;

  v8f acc0 = {}; v8f acc1 = {};
  #pragma unroll
  for (int kp = 0; kp < 4; ++kp) {
    v16bf a0 = fragA_from_f32(xrow, 2 * kp,     half);
    v16bf a1 = fragA_from_f32(xrow, 2 * kp + 1, half);
    acc0 = __builtin_amdgcn_wmma_f32_16x16x32_bf16(false, a0, false, B[2*kp],
                                                   (short)0, acc0, false, false);
    acc1 = __builtin_amdgcn_wmma_f32_16x16x32_bf16(false, a1, false, B[2*kp+1],
                                                   (short)0, acc1, false, false);
  }

  const float bias = b_ih[col];
  #pragma unroll
  for (int r = 0; r < 8; ++r) {
    const int m = half * 8 + r;               // C/D layout: m = r + 8*(lane>=16)
    pre[(size_t)(mt * 16 + m) * RNN_H + col] = acc0[r] + acc1[r] + bias;
  }
}

// ---- Phase 2: fused bidirectional recurrence -------------------------------
// One workgroup, 16 waves. Batch rows 0..7 = forward stream, 8..15 = reverse
// stream -> M=16 exactly fills the WMMA M dimension; both directions advance
// in the same matrix op. W_hh^T stays in VGPRs; h ping-pongs through LDS.
__global__ __launch_bounds__(512)
void rnn_recur_wmma(const float* __restrict__ pre,
                    const float* __restrict__ W_hh,
                    const float* __restrict__ b_hh,
                    const float* __restrict__ init_states,
                    float* __restrict__ out) {
  __shared__ __align__(16) bf16_t hbuf[2][16][HPAD];

  const int tid  = threadIdx.x;
  const int lane = tid & 31;
  const int wave = tid >> 5;                 // 0..15 -> N-tile of H
  const int half = lane >> 4;
  const int l16  = lane & 15;
  const int col  = wave * 16 + l16;          // global hidden index n

  // h0 = broadcast(init_states) over all 16 batch rows
  for (int i = tid; i < 16 * RNN_H; i += 512) {
    const int rr = i / RNN_H, cc = i % RNN_H;
    hbuf[0][rr][cc] = (bf16_t)init_states[cc];
  }

  // Resident weights: B[k][n] = W_hh[n][k] -> row `col` of W_hh (64 VGPRs/lane)
  v16bf B[8];
  #pragma unroll
  for (int kc = 0; kc < 8; ++kc)
    B[kc] = fragB_from_f32(W_hh + (size_t)col * RNN_H, kc, half);

  const float bias = b_hh[col];
  __syncthreads();

  int pp = 0;
  for (int t = 0; t < RNN_L; ++t) {
    const int tt = half ? (RNN_L - 1 - t) : t;   // reverse rows walk flipped time
    // Prefetch pre_t for this step's 8 output rows (nb == r for both halves);
    // issued before the WMMAs so L2 latency hides under the matrix ops.
    float pv[8];
    #pragma unroll
    for (int r = 0; r < 8; ++r)
      pv[r] = pre[((size_t)r * RNN_L + tt) * RNN_H + col];

    const bf16_t* hrow = &hbuf[pp][l16][0];
    v8f acc0 = {}; v8f acc1 = {};
    #pragma unroll
    for (int kp = 0; kp < 4; ++kp) {
      v16bf a0 = fragA_from_lds(hrow, 2 * kp,     half);
      v16bf a1 = fragA_from_lds(hrow, 2 * kp + 1, half);
      acc0 = __builtin_amdgcn_wmma_f32_16x16x32_bf16(false, a0, false, B[2*kp],
                                                     (short)0, acc0, false, false);
      acc1 = __builtin_amdgcn_wmma_f32_16x16x32_bf16(false, a1, false, B[2*kp+1],
                                                     (short)0, acc1, false, false);
    }

    const size_t outc = half ? (size_t)(RNN_H + col) : (size_t)col;
    #pragma unroll
    for (int r = 0; r < 8; ++r) {
      const float hn = tanhf(acc0[r] + acc1[r] + pv[r] + bias);
      out[((size_t)r * RNN_L + tt) * (2 * RNN_H) + outc] = hn;
      hbuf[pp ^ 1][half * 8 + r][col] = (bf16_t)hn;
    }
    __syncthreads();   // publish h_{t+1} before any wave reads it
    pp ^= 1;
  }
}

// ---- Host-side launch ------------------------------------------------------
extern "C" void kernel_launch(void* const* d_in, const int* in_sizes, int n_in,
                              void* d_out, int out_size, void* d_ws, size_t ws_size,
                              hipStream_t stream) {
  const float* x           = (const float*)d_in[0];  // (N, L, C)
  const float* W_ih        = (const float*)d_in[1];  // (H, C)
  const float* W_hh        = (const float*)d_in[2];  // (H, H)
  const float* b_ih        = (const float*)d_in[3];  // (H,)
  const float* b_hh        = (const float*)d_in[4];  // (H,)
  const float* init_states = (const float*)d_in[5];  // (H,)
  float* out = (float*)d_out;                        // (N, L, 2H)
  float* pre = (float*)d_ws;                         // (N, L, H) f32 = 32 MB scratch

  // Phase 1: saturate the chip with the input-projection GEMM (shared by both
  // directions since proj(flip(x)) == flip(proj(x))).
  dim3 g1((RNN_N * RNN_L / 16) / 8, RNN_H / 16);     // (256, 16) blocks x 256 thr
  rnn_inproj_wmma<<<g1, 256, 0, stream>>>(x, W_ih, b_ih, pre);

  // Phase 2: latency-bound fused bidirectional recurrence, one WGP.
  rnn_recur_wmma<<<1, 512, 0, stream>>>(pre, W_hh, b_hh, init_states, out);
}
